// RNN_28415503631005
// MI455X (gfx1250) — matrix operand order repaired and checked by reading, not compiled, not run on hardware
//
#include <hip/hip_runtime.h>

// ---------------------------------------------------------------------------
// 2-layer tanh RNN, T=2048, B=64, I=H=512 for MI455X (gfx1250, wave32, WMMA)
//
// Stage 1 (prep):    convert weights f32->bf16 into WMMA-B-swizzled layout,
//                    combine biases.
// Stage 2 (pregemm): Z0 = x @ W_ih0 + b0 for all T in parallel (bf16 WMMA),
//                    written into d_out's [T,B,H] region (consumed in-place).
// Stage 3 (rnn):     4 independent persistent workgroups (one per 16-row
//                    batch tile), 16 waves x 2 N-tiles each; h0/h1 state in
//                    LDS as bf16 (double buffered); per step:
//                      h0' = tanh(Z0[t] + h0 @ W_hh0)        (WMMA, K=512)
//                      h1' = tanh(h0' @ W_ih1 + h1 @ W_hh1 + b1)
//                    only __syncthreads() (s_barrier) — no global sync.
// ---------------------------------------------------------------------------

typedef __attribute__((ext_vector_type(16))) __bf16 v16bf;
typedef __attribute__((ext_vector_type(8)))  __bf16 v8bf;
typedef __attribute__((ext_vector_type(8)))  float  v8f;
typedef __attribute__((ext_vector_type(4)))  float  v4f;

#define T_STEPS 2048
#define BATCH   64
#define HID     512
#define NT      (HID / 16)   // 32 column tiles
#define KC      (HID / 32)   // 16 K-chunks of 32
#define LDSTR   520          // padded LDS row stride (bf16) -> conflict-free
#define NTW     2            // N-tiles per wave in the recurrence
#define RWAVES  (NT / NTW)   // 16 waves
#define RTHREADS (RWAVES * 32)

// --------------------------------------------------------------------------
// B-fragment swizzle for V_WMMA_F32_16X16X32_BF16:
//   element (K=k, N=n) -> lane = ((k>>3)&1)*16 + (n&15), e = (k&7) | (((k&31)>>4)<<3)
//   linear chunks of (ntile, kc) so a wave loads 16 contiguous bf16 per lane.
// --------------------------------------------------------------------------
__device__ __host__ __forceinline__ size_t bswz(int k, int n) {
  int ntile = n >> 4, nin = n & 15;
  int kc = k >> 5,  kk = k & 31;
  int hi = (kk >> 3) & 1, e = (kk & 7) | ((kk >> 4) << 3);
  int lane = hi * 16 + nin;
  return ((size_t)(ntile * KC + kc) * 32 + lane) * 16 + e;
}

__device__ __forceinline__ v16bf lds_afrag(const __bf16* arow, int kc, int hi) {
  const int c0 = kc * 32 + hi * 8;
  v8bf alo = *(const v8bf*)(arow + c0);
  v8bf ahi = *(const v8bf*)(arow + c0 + 16);
  v16bf a = {};
  for (int j = 0; j < 8; ++j) { a[j] = alo[j]; a[8 + j] = ahi[j]; }
  return a;
}

__device__ __forceinline__ v16bf bfrag(const __bf16* wsw, int ntile, int kc, int lane) {
  return *(const v16bf*)(wsw + ((size_t)(ntile * KC + kc) * 32 + lane) * 16);
}

// ---------------------------------------------------------------------------
// Stage 1: weight conversion + swizzle, bias combine
// ---------------------------------------------------------------------------
__global__ __launch_bounds__(256)
void rnn_prep_kernel(const float* __restrict__ wih0, const float* __restrict__ whh0,
                     const float* __restrict__ wih1, const float* __restrict__ whh1,
                     const float* __restrict__ bih0, const float* __restrict__ bhh0,
                     const float* __restrict__ bih1, const float* __restrict__ bhh1,
                     __bf16* __restrict__ sw_ih0, __bf16* __restrict__ sw_hh0,
                     __bf16* __restrict__ sw_ih1, __bf16* __restrict__ sw_hh1,
                     float* __restrict__ b0c, float* __restrict__ b1c) {
  int idx = blockIdx.x * blockDim.x + threadIdx.x;
  if (idx < HID * HID) {
    int k = idx >> 9, n = idx & (HID - 1);
    size_t d = bswz(k, n);
    sw_ih0[d] = (__bf16)wih0[idx];
    sw_hh0[d] = (__bf16)whh0[idx];
    sw_ih1[d] = (__bf16)wih1[idx];
    sw_hh1[d] = (__bf16)whh1[idx];
  }
  if (idx < HID) {
    b0c[idx] = bih0[idx] + bhh0[idx];
    b1c[idx] = bih1[idx] + bhh1[idx];
  }
}

// ---------------------------------------------------------------------------
// Stage 2: Z0[t,b,n] = x[t,b,:] @ W_ih0 + b0   (M = T*B = 131072)
// One block per 16-row M-tile; 8 waves x 4 N-tiles each.
// A-fragment per ISA 16-bit A layout: lane(hi) -> K = kc*32+hi*8+{0..7,16..23}
// ---------------------------------------------------------------------------
__global__ __launch_bounds__(256)
void rnn_pregemm_kernel(const float* __restrict__ x,
                        const __bf16* __restrict__ wsw,
                        const float* __restrict__ b0c,
                        float* __restrict__ z) {
  const int mtile = blockIdx.x;            // 0..8191
  const int wv    = threadIdx.x >> 5;      // 0..7
  const int lane  = threadIdx.x & 31;
  const int lrow  = lane & 15;
  const int hi    = lane >> 4;
  const float* arow = x + ((size_t)mtile * 16 + lrow) * HID;

  v8f acc[4];
  for (int i = 0; i < 4; ++i) {
    float bb = b0c[(wv * 4 + i) * 16 + lrow];
    for (int r = 0; r < 8; ++r) acc[i][r] = bb;
  }

  for (int kc = 0; kc < KC; ++kc) {
    const int c0 = kc * 32 + hi * 8;
    v4f f0 = *(const v4f*)(arow + c0);
    v4f f1 = *(const v4f*)(arow + c0 + 4);
    v4f f2 = *(const v4f*)(arow + c0 + 16);
    v4f f3 = *(const v4f*)(arow + c0 + 20);
    v16bf a = {};
    for (int j = 0; j < 4; ++j) {
      a[j]      = (__bf16)f0[j];
      a[4 + j]  = (__bf16)f1[j];
      a[8 + j]  = (__bf16)f2[j];
      a[12 + j] = (__bf16)f3[j];
    }
    for (int i = 0; i < 4; ++i) {
      const v16bf b = bfrag(wsw, wv * 4 + i, kc, lane);
      acc[i] = __builtin_amdgcn_wmma_f32_16x16x32_bf16(
          false, a, false, b, (short)0, acc[i], false, false);
    }
  }

  for (int i = 0; i < 4; ++i) {
    const int n0 = (wv * 4 + i) * 16;
    for (int r = 0; r < 8; ++r) {
      const size_t row = (size_t)mtile * 16 + hi * 8 + r;
      z[row * HID + n0 + lrow] = acc[i][r];
    }
  }
}

// ---------------------------------------------------------------------------
// Stage 3: persistent recurrence. 4 blocks (one per batch 16-row tile),
// 512 threads = 16 waves, wave w owns output columns [32w, 32w+32)
// (two 16x16 WMMA N-tiles sharing one A fragment per K-chunk).
// h0/h1 double-buffered bf16 in LDS; 2 s_barriers per step.
// ---------------------------------------------------------------------------
__global__ __launch_bounds__(RTHREADS)
void rnn_recurrence_kernel(const __bf16* __restrict__ sw_hh0,
                           const __bf16* __restrict__ sw_ih1,
                           const __bf16* __restrict__ sw_hh1,
                           const float* __restrict__ b1c,
                           float* __restrict__ out) {
  extern __shared__ char smem[];
  __bf16* h0buf = (__bf16*)smem;                  // [2][16][LDSTR]
  __bf16* h1buf = h0buf + 2 * 16 * LDSTR;         // [2][16][LDSTR]

  const int mtile = blockIdx.x;                   // 0..3
  const int wv    = threadIdx.x >> 5;             // 0..15
  const int lane  = threadIdx.x & 31;
  const int lrow  = lane & 15;
  const int hi    = lane >> 4;

  float* zys     = out;                                     // [T][B][H]
  float* hiddens = out + (size_t)T_STEPS * BATCH * HID;     // [2][B][H]

  // zero h0/h1
  for (int i = threadIdx.x; i < 4 * 16 * LDSTR; i += RTHREADS)
    h0buf[i] = (__bf16)0.0f;
  __syncthreads();

  float bias1[NTW];
  for (int i = 0; i < NTW; ++i) bias1[i] = b1c[(wv * NTW + i) * 16 + lrow];

  int p = 0;

  for (int t = 0; t < T_STEPS; ++t) {
    const __bf16* h0r = h0buf + p * 16 * LDSTR;
    __bf16*       h0w = h0buf + (1 - p) * 16 * LDSTR;
    const __bf16* h1r = h1buf + p * 16 * LDSTR;
    __bf16*       h1w = h1buf + (1 - p) * 16 * LDSTR;

    const float* zp = zys + ((size_t)t * BATCH + mtile * 16) * HID;

    // Pull next step's Z0 tile toward this WGP while we compute.
    if (t + 1 < T_STEPS)
      __builtin_prefetch(zp + (size_t)BATCH * HID +
                             (size_t)(hi * 8) * HID + (wv * NTW) * 16 + lrow, 0, 1);

    // ---------- layer 0: h0' = tanh(Z0[t] + h0 @ W_hh0) ----------
    v8f acc[NTW];
    for (int i = 0; i < NTW; ++i)
      for (int r = 0; r < 8; ++r)
        acc[i][r] = zp[(size_t)(hi * 8 + r) * HID + (wv * NTW + i) * 16 + lrow];

    {
      const __bf16* arow = h0r + lrow * LDSTR;
#pragma unroll 4
      for (int kc = 0; kc < KC; ++kc) {
        const v16bf a = lds_afrag(arow, kc, hi);
        for (int i = 0; i < NTW; ++i) {
          const v16bf b = bfrag(sw_hh0, wv * NTW + i, kc, lane);
          acc[i] = __builtin_amdgcn_wmma_f32_16x16x32_bf16(
              false, a, false, b, (short)0, acc[i], false, false);
        }
      }
    }
    float h0v[NTW][8];
    for (int i = 0; i < NTW; ++i)
      for (int r = 0; r < 8; ++r) h0v[i][r] = tanhf(acc[i][r]);
    for (int i = 0; i < NTW; ++i)
      for (int r = 0; r < 8; ++r)
        h0w[(hi * 8 + r) * LDSTR + (wv * NTW + i) * 16 + lrow] = (__bf16)h0v[i][r];
    if (t == T_STEPS - 1) {
      for (int i = 0; i < NTW; ++i)
        for (int r = 0; r < 8; ++r)
          hiddens[(size_t)(mtile * 16 + hi * 8 + r) * HID + (wv * NTW + i) * 16 + lrow] = h0v[i][r];
    }
    __syncthreads();   // h0' visible to all waves

    // ---------- layer 1: h1' = tanh(h0' @ W_ih1 + h1 @ W_hh1 + b1) ----------
    // Single merged K-loop: two A fragments (h0', h1) feed 4 WMMAs per chunk.
    for (int i = 0; i < NTW; ++i)
      for (int r = 0; r < 8; ++r) acc[i][r] = bias1[i];
    {
      const __bf16* arow0 = h0w + lrow * LDSTR;   // fresh h0'
      const __bf16* arow1 = h1r + lrow * LDSTR;   // previous h1
#pragma unroll 4
      for (int kc = 0; kc < KC; ++kc) {
        const v16bf a0 = lds_afrag(arow0, kc, hi);
        const v16bf a1 = lds_afrag(arow1, kc, hi);
        for (int i = 0; i < NTW; ++i) {
          const v16bf b0 = bfrag(sw_ih1, wv * NTW + i, kc, lane);
          acc[i] = __builtin_amdgcn_wmma_f32_16x16x32_bf16(
              false, a0, false, b0, (short)0, acc[i], false, false);
          const v16bf b1 = bfrag(sw_hh1, wv * NTW + i, kc, lane);
          acc[i] = __builtin_amdgcn_wmma_f32_16x16x32_bf16(
              false, a1, false, b1, (short)0, acc[i], false, false);
        }
      }
    }
    float h1v[NTW][8];
    for (int i = 0; i < NTW; ++i)
      for (int r = 0; r < 8; ++r) h1v[i][r] = tanhf(acc[i][r]);
    for (int i = 0; i < NTW; ++i)
      for (int r = 0; r < 8; ++r)
        h1w[(hi * 8 + r) * LDSTR + (wv * NTW + i) * 16 + lrow] = (__bf16)h1v[i][r];
    {
      float* yp = zys + ((size_t)t * BATCH + mtile * 16) * HID;
      for (int i = 0; i < NTW; ++i)
        for (int r = 0; r < 8; ++r)
          yp[(size_t)(hi * 8 + r) * HID + (wv * NTW + i) * 16 + lrow] = h1v[i][r]; // overwrites Z0[t]
    }
    if (t == T_STEPS - 1) {
      for (int i = 0; i < NTW; ++i)
        for (int r = 0; r < 8; ++r)
          hiddens[(size_t)(BATCH + mtile * 16 + hi * 8 + r) * HID + (wv * NTW + i) * 16 + lrow] = h1v[i][r];
    }
    __syncthreads();   // h1' done; safe to recycle buffers
    p ^= 1;
  }
}

// ---------------------------------------------------------------------------
extern "C" void kernel_launch(void* const* d_in, const int* in_sizes, int n_in,
                              void* d_out, int out_size, void* d_ws, size_t ws_size,
                              hipStream_t stream) {
  const float* x    = (const float*)d_in[0];
  const float* wih0 = (const float*)d_in[1];
  const float* whh0 = (const float*)d_in[2];
  const float* bih0 = (const float*)d_in[3];
  const float* bhh0 = (const float*)d_in[4];
  const float* wih1 = (const float*)d_in[5];
  const float* whh1 = (const float*)d_in[6];
  const float* bih1 = (const float*)d_in[7];
  const float* bhh1 = (const float*)d_in[8];
  float* out = (float*)d_out;

  const size_t WN = (size_t)HID * HID;       // 262144
  __bf16* sw_ih0 = (__bf16*)d_ws;
  __bf16* sw_hh0 = sw_ih0 + WN;
  __bf16* sw_ih1 = sw_hh0 + WN;
  __bf16* sw_hh1 = sw_ih1 + WN;
  float*  b0c    = (float*)(sw_hh1 + WN);
  float*  b1c    = b0c + HID;

  // Stage 1: weight swizzle/convert + bias combine
  rnn_prep_kernel<<<(int)((WN + 255) / 256), 256, 0, stream>>>(
      wih0, whh0, wih1, whh1, bih0, bhh0, bih1, bhh1,
      sw_ih0, sw_hh0, sw_ih1, sw_hh1, b0c, b1c);

  // Stage 2: Z0 = x @ W_ih0 + b0, into d_out's [T,B,H] region
  rnn_pregemm_kernel<<<(T_STEPS * BATCH) / 16, 256, 0, stream>>>(
      x, sw_ih0, b0c, out);

  // Stage 3: recurrence (4 independent batch tiles, LDS-resident state)
  const size_t lds_bytes = 4 * 16 * LDSTR * sizeof(__bf16);  // 66560 B
  rnn_recurrence_kernel<<<4, RTHREADS, lds_bytes, stream>>>(
      sw_hh0, sw_ih1, sw_hh1, b1c, out);
}
